// ErnieLayoutSelfAttention_9234179687493
// MI455X (gfx1250) — matrix operand order repaired and checked by reading, not compiled, not run on hardware
//
#include <hip/hip_runtime.h>
#include <hip/hip_bf16.h>
#include <cstdint>

// ---------------------------------------------------------------------------
// ErnieLayout self-attention for gfx1250 (MI455X), wave32 + WMMA bf16 + TDM.
//   B=2, S=2048, H=768, NH=12, HD=64
// Pipeline:
//   1) cvt_hidden:     f32 hidden -> bf16           [4096 x 768]
//   2) cvt_w_tr (x3):  f32 W[k,n] -> bf16 Wt[n,k]   [768 x 768]
//   3) qkv_kernel:     GEMM (WMMA bf16) + bias + RoPE (+1/8 scale folded in Q)
//                      Q,K -> [b,h,s,64] bf16 ; V -> transposed [b,h,64,s] bf16
//   4) attn_kernel:    flash attention; K/V tiles staged to LDS once per block
//                      via the Tensor Data Mover (double-buffered, TENSORcnt),
//                      WMMA for QK^T and PV, online softmax,
//                      rel_pos + rel_2d_pos + mask streamed once (HBM-bound).
// ---------------------------------------------------------------------------

#define NB   2
#define SQ   2048
#define HID  768
#define NH   12
#define HD   64

typedef __attribute__((ext_vector_type(16))) __bf16 v16bf;
typedef __attribute__((ext_vector_type(8)))  float  v8f;
typedef unsigned int u32x4 __attribute__((ext_vector_type(4)));
typedef int          i32x8 __attribute__((ext_vector_type(8)));
typedef int          i32x4 __attribute__((ext_vector_type(4)));

#if defined(__has_builtin)
#if __has_builtin(__builtin_amdgcn_tensor_load_to_lds) && \
    __has_builtin(__builtin_amdgcn_s_wait_tensorcnt)
#define ATHENA_USE_TDM 1
#endif
#endif

union Frag {
  v16bf v;
  uint4 u[2];
  unsigned short s[16];
};

__device__ __forceinline__ unsigned short f2bf(float f) {
  unsigned int u = __float_as_uint(f);
  u += 0x7FFFu + ((u >> 16) & 1u);          // round to nearest even
  return (unsigned short)(u >> 16);
}

__device__ __forceinline__ v8f wmma_bf16(const Frag& a, const Frag& b, v8f c) {
  return __builtin_amdgcn_wmma_f32_16x16x32_bf16(
      false, a.v, false, b.v, (short)0, c, false, false);
}

#ifdef ATHENA_USE_TDM
// Tensor DMA: load a 2D tile (tile_w x tile_h elements, 2-byte elements) from
// a row-major tensor (row stride = row_stride elems) into contiguous LDS.
// D# packing per CDNA5 ISA 8.3/8.4 (group0: count/lds/global/type=2;
// group1: data_size, tensor dims, tile dims, dim0 stride).
// This toolchain exposes the 6-arg builtin:
//   (uint32x4 g0, int32x8 g1, int32x4 g2, int32x4 g3, int32x8 g4, i32 cpol)
__device__ __forceinline__ void tdm_load_2d(void* lds_ptr, const void* gptr,
                                            unsigned tile_w, unsigned tile_h,
                                            unsigned tensor_w, unsigned tensor_h,
                                            unsigned row_stride) {
  const unsigned lds_off = (unsigned)(uintptr_t)lds_ptr;   // addr[31:0] = LDS
  const unsigned long long ga = (unsigned long long)(uintptr_t)gptr;
  u32x4 g0;
  g0.x = 1u;                                   // count=1, user descriptor
  g0.y = lds_off;                              // lds_addr
  g0.z = (unsigned)ga;                         // global_addr[31:0]
  g0.w = (unsigned)(ga >> 32) | (2u << 30);    // global_addr[56:32] | type=2
  i32x8 g1;
  g1[0] = (int)(1u << 16);                                  // data_size = 2B
  g1[1] = (int)((tensor_w & 0xFFFFu) << 16);                // tensor_dim0 lo
  g1[2] = (int)((tensor_w >> 16) | ((tensor_h & 0xFFFFu) << 16));
  g1[3] = (int)((tensor_h >> 16) | ((tile_w & 0xFFFFu) << 16)); // tile_dim0
  g1[4] = (int)(tile_h & 0xFFFFu);                          // tile_dim1
  g1[5] = (int)row_stride;                                  // dim0 stride lo32
  g1[6] = 0;
  g1[7] = 0;
  i32x4 g2 = {1, 0, 0, 0};                                  // tensor_dim2 = 1
  i32x4 g3 = {0, 0, 0, 0};
  i32x8 g4 = {0, 0, 0, 0, 0, 0, 0, 0};
  __builtin_amdgcn_tensor_load_to_lds(g0, g1, g2, g3, g4, 0);
}
#endif

// ---------------------------------------------------------------------------
// 1) hidden f32 -> bf16 (element-wise)
// ---------------------------------------------------------------------------
__global__ void cvt_hidden(const float* __restrict__ src,
                           unsigned short* __restrict__ dst, int n) {
  int i = blockIdx.x * 256 + threadIdx.x;
  if (i < n) dst[i] = f2bf(src[i]);
}

// ---------------------------------------------------------------------------
// 2) W[k*768+n] f32 -> Wt[n*768+k] bf16  (tiled transpose through LDS)
// ---------------------------------------------------------------------------
__global__ void cvt_w_tr(const float* __restrict__ w,
                         unsigned short* __restrict__ wt) {
  __shared__ float tile[32][33];
  int n0 = blockIdx.x * 32, k0 = blockIdx.y * 32;
  int tx = threadIdx.x & 31, ty = threadIdx.x >> 5;   // 32 x 8
  #pragma unroll
  for (int r = ty; r < 32; r += 8) tile[r][tx] = w[(k0 + r) * HID + n0 + tx];
  __syncthreads();
  #pragma unroll
  for (int r = ty; r < 32; r += 8)
    wt[(size_t)(n0 + r) * HID + k0 + tx] = f2bf(tile[tx][r]);
}

// ---------------------------------------------------------------------------
// 3) QKV projection + bias + RoPE. grid = (12 heads, 32, 3 mats), block = 256.
//    Each wave owns a 16x64 (rows x features-of-one-head) output tile.
// ---------------------------------------------------------------------------
__global__ __launch_bounds__(256) void qkv_kernel(
    const unsigned short* __restrict__ hid,   // [4096,768] bf16
    const unsigned short* __restrict__ wqT,
    const unsigned short* __restrict__ wkT,
    const unsigned short* __restrict__ wvT,   // [768(n),768(k)] bf16
    const float* __restrict__ bq, const float* __restrict__ bk,
    const float* __restrict__ bv,
    unsigned short* __restrict__ qbf,         // [B,NH,S,HD]
    unsigned short* __restrict__ kbf,         // [B,NH,S,HD]
    unsigned short* __restrict__ vtb) {       // [B,NH,HD,S]
  const int z = blockIdx.z;
  const unsigned short* wT = (z == 0) ? wqT : (z == 1) ? wkT : wvT;
  const float* bias = (z == 0) ? bq : (z == 1) ? bk : bv;

  const int lane = threadIdx.x & 31;
  const int wv_id = threadIdx.x >> 5;
  const int ln = lane & 15, half = lane >> 4;
  const int m0 = (blockIdx.y * 8 + wv_id) * 16;   // row tile (b*S+s space)
  const int h  = blockIdx.x;                      // head
  const int n0 = h * HD;
  const int mrow = m0 + ln;

  v8f c[4] = {{}, {}, {}, {}};

  for (int kk = 0; kk < HID; kk += 32) {
    Frag a;
    const unsigned short* ap = hid + (size_t)mrow * HID + kk + half * 8;
    a.u[0] = *(const uint4*)(ap);
    a.u[1] = *(const uint4*)(ap + 16);
    #pragma unroll
    for (int j = 0; j < 4; ++j) {
      Frag bmat;
      const unsigned short* bp =
          wT + (size_t)(n0 + j * 16 + ln) * HID + kk + half * 8;
      bmat.u[0] = *(const uint4*)(bp);
      bmat.u[1] = *(const uint4*)(bp + 16);
      c[j] = wmma_bf16(a, bmat, c[j]);
    }
  }

  float bval[4];
  #pragma unroll
  for (int j = 0; j < 4; ++j) bval[j] = bias[n0 + j * 16 + ln];

  if (z < 2) {
    // RoPE: d and d+32 sit in accumulator pair (c[jp], c[jp+2]) same position.
    unsigned short* outp = (z == 0) ? qbf : kbf;
    const float scale = (z == 0) ? 0.125f : 1.0f;   // 1/sqrt(64) folded into Q
    #pragma unroll
    for (int jp = 0; jp < 2; ++jp) {
      const int d_lo = jp * 16 + ln;                 // [0,32)
      const float freq = __powf(10000.0f, -(float)(2 * d_lo) * (1.0f / 64.0f));
      #pragma unroll
      for (int r = 0; r < 8; ++r) {
        const int mr = m0 + r + half * 8;
        const int s  = mr & (SQ - 1);
        const int bb = mr >> 11;
        float sn, cs;
        __sincosf((float)s * freq, &sn, &cs);
        const float xlo = c[jp][r] + bval[jp];
        const float xhi = c[jp + 2][r] + bval[jp + 2];
        unsigned short* base =
            outp + (((size_t)(bb * NH + h) * SQ + s) * HD);
        base[d_lo]      = f2bf((xlo * cs - xhi * sn) * scale);
        base[d_lo + 32] = f2bf((xhi * cs + xlo * sn) * scale);
      }
    }
  } else {
    // V: store transposed [b,h,d,s] so PV B-fragments are contiguous in s.
    #pragma unroll
    for (int j = 0; j < 4; ++j) {
      const int d = j * 16 + ln;
      #pragma unroll
      for (int r = 0; r < 8; ++r) {
        const int mr = m0 + r + half * 8;
        const int s  = mr & (SQ - 1);
        const int bb = mr >> 11;
        vtb[((size_t)(bb * NH + h) * HD + d) * SQ + s] = f2bf(c[j][r] + bval[j]);
      }
    }
  }
}

// ---------------------------------------------------------------------------
// 4) Flash attention. grid = (S/64, NH, B), block = 128 (4 waves).
//    Wave w owns queries [q0, q0+16); loops over keys in chunks of 64.
//    K/V 64-key tiles are staged into LDS once per block (TDM, double-buffered)
//    and shared by all 4 waves.
// ---------------------------------------------------------------------------
__global__ __launch_bounds__(128) void attn_kernel(
    const unsigned short* __restrict__ qbf,   // [B,NH,S,HD] (prescaled 1/8)
    const unsigned short* __restrict__ kbf,   // [B,NH,S,HD]
    const unsigned short* __restrict__ vtb,   // [B,NH,HD,S]
    const float* __restrict__ rel,            // [B,NH,S,S]
    const float* __restrict__ rel2,           // [B,NH,S,S]
    const float* __restrict__ mask,           // [B,1,1,S]
    float* __restrict__ out) {                // [B,S,H]
  __shared__ unsigned short Kbuf[2][64 * HD]; // 2 x 8KB  (keys x dims)
  __shared__ unsigned short Vbuf[2][64 * HD]; // 2 x 8KB  (dims x keys)
  __shared__ unsigned short P[4][16 * 64];    // per-wave probs tile (bf16)

  const int lane = threadIdx.x & 31;
  const int wv_id = threadIdx.x >> 5;
  const int ln = lane & 15, half = lane >> 4;
  const int b = blockIdx.z, h = blockIdx.y;
  const int q0 = blockIdx.x * 64 + wv_id * 16;

  const size_t bh = (size_t)(b * NH + h);
  const unsigned short* Q  = qbf + bh * SQ * HD;
  const unsigned short* K  = kbf + bh * SQ * HD;
  const unsigned short* VT = vtb + bh * HD * SQ;
  const float* relp  = rel  + bh * SQ * SQ;
  const float* rel2p = rel2 + bh * SQ * SQ;
  const float* mk    = mask + (size_t)b * SQ;

#ifdef ATHENA_USE_TDM
  if (wv_id == 0) {   // stage chunk 0 (K: 64x64 contiguous; V: 64 rows, stride S)
    tdm_load_2d(&Kbuf[0][0], K, 64 * HD, 1, 64 * HD, 1, 64 * HD);
    tdm_load_2d(&Vbuf[0][0], VT, 64, HD, SQ, HD, SQ);
  }
#endif

  // Q A-fragments (K-dim 0..31 and 32..63), loaded once.
  Frag aq0, aq1;
  {
    const unsigned short* qp = Q + (size_t)(q0 + ln) * HD;
    aq0.u[0] = *(const uint4*)(qp + half * 8);
    aq0.u[1] = *(const uint4*)(qp + 16 + half * 8);
    aq1.u[0] = *(const uint4*)(qp + 32 + half * 8);
    aq1.u[1] = *(const uint4*)(qp + 48 + half * 8);
  }

  float mstat[8], lstat[8];
  v8f ctx[4] = {{}, {}, {}, {}};
  #pragma unroll
  for (int r = 0; r < 8; ++r) { mstat[r] = -1e30f; lstat[r] = 0.0f; }

  for (int kb = 0; kb < SQ; kb += 64) {
    const int p = (kb >> 6) & 1;
    const unsigned short* kc_ = &Kbuf[p][0];
    const unsigned short* vc_ = &Vbuf[p][0];

#ifdef ATHENA_USE_TDM
    if (wv_id == 0) {
      if (kb + 64 < SQ) {   // overlap copy of next chunk with compute
        tdm_load_2d(&Kbuf[p ^ 1][0], K + (size_t)(kb + 64) * HD,
                    64 * HD, 1, 64 * HD, 1, 64 * HD);
        tdm_load_2d(&Vbuf[p ^ 1][0], VT + kb + 64, 64, HD, SQ, HD, SQ);
        __builtin_amdgcn_s_wait_tensorcnt(2);   // chunk kb complete (in-order)
      } else {
        __builtin_amdgcn_s_wait_tensorcnt(0);
      }
    }
    __syncthreads();        // publish chunk kb to all waves
#else
    {   // cooperative fallback staging of the current chunk
      const int tid = threadIdx.x;
      uint4* lk = (uint4*)&Kbuf[p][0];
      const uint4* gk = (const uint4*)(K + (size_t)kb * HD);
      uint4* lv = (uint4*)&Vbuf[p][0];
      #pragma unroll
      for (int t = 0; t < 4; ++t) {
        const int idx = tid + 128 * t;            // 512 uint4 per buffer
        lk[idx] = gk[idx];
        const int r = idx >> 3, c0 = idx & 7;
        lv[idx] = *(const uint4*)(VT + (size_t)r * SQ + kb + c0 * 8);
      }
    }
    __syncthreads();
#endif

    // prefetch next chunk of the dominant stream (rel_pos)
    if (kb + 64 < SQ)
      __builtin_prefetch(relp + (size_t)(q0 + half * 8) * SQ + kb + 64 + ln * 4,
                         0, 1);

    // ---- scores: 4 x (16q x 16k) tiles (K fragments from LDS) ------------
    v8f st[4];
    #pragma unroll
    for (int t = 0; t < 4; ++t) {
      const int key16 = kb + t * 16;
      Frag bk0, bk1;
      const unsigned short* kp = kc_ + (t * 16 + ln) * HD + half * 8;
      bk0.u[0] = *(const uint4*)(kp);
      bk0.u[1] = *(const uint4*)(kp + 16);
      bk1.u[0] = *(const uint4*)(kp + 32);
      bk1.u[1] = *(const uint4*)(kp + 48);
      v8f acc = {};
      acc = wmma_bf16(aq0, bk0, acc);
      acc = wmma_bf16(aq1, bk1, acc);
      const float maskv = mk[key16 + ln];
      #pragma unroll
      for (int r = 0; r < 8; ++r) {
        const size_t off = (size_t)(q0 + r + half * 8) * SQ + key16 + ln;
        acc[r] += relp[off] + rel2p[off] + maskv;
      }
      st[t] = acc;
    }

    // ---- online softmax (row = register r, 16 cols across lane half) -----
    #pragma unroll
    for (int r = 0; r < 8; ++r) {
      float cm = fmaxf(fmaxf(st[0][r], st[1][r]), fmaxf(st[2][r], st[3][r]));
      #pragma unroll
      for (int xm = 1; xm < 16; xm <<= 1) cm = fmaxf(cm, __shfl_xor(cm, xm, 32));
      const float mn = fmaxf(mstat[r], cm);
      const float alpha = __expf(mstat[r] - mn);
      mstat[r] = mn;
      float rs = 0.0f;
      #pragma unroll
      for (int t = 0; t < 4; ++t) {
        const float p2 = __expf(st[t][r] - mn);
        st[t][r] = p2;
        rs += p2;
      }
      #pragma unroll
      for (int xm = 1; xm < 16; xm <<= 1) rs += __shfl_xor(rs, xm, 32);
      lstat[r] = lstat[r] * alpha + rs;
      #pragma unroll
      for (int j = 0; j < 4; ++j) ctx[j][r] *= alpha;
    }

    // ---- reshape P: C-layout -> A-layout via per-wave LDS tile -----------
    // (same-wave DS ops are in-order on CDNA5; no block barrier needed here)
    unsigned short* pw = &P[wv_id][0];
    #pragma unroll
    for (int t = 0; t < 4; ++t)
      #pragma unroll
      for (int r = 0; r < 8; ++r)
        pw[(r + half * 8) * 64 + t * 16 + ln] = f2bf(st[t][r]);

    Frag ap0, ap1;
    {
      const unsigned short* pr = pw + ln * 64;
      ap0.u[0] = *(const uint4*)(pr + half * 8);
      ap0.u[1] = *(const uint4*)(pr + 16 + half * 8);
      ap1.u[0] = *(const uint4*)(pr + 32 + half * 8);
      ap1.u[1] = *(const uint4*)(pr + 48 + half * 8);
    }

    // ---- ctx += P @ V (V fragments from LDS) -----------------------------
    #pragma unroll
    for (int j = 0; j < 4; ++j) {
      const unsigned short* vp = vc_ + (j * 16 + ln) * 64 + half * 8;
      Frag bv0, bv1;
      bv0.u[0] = *(const uint4*)(vp);
      bv0.u[1] = *(const uint4*)(vp + 16);
      bv1.u[0] = *(const uint4*)(vp + 32);
      bv1.u[1] = *(const uint4*)(vp + 48);
      ctx[j] = wmma_bf16(ap0, bv0, ctx[j]);
      ctx[j] = wmma_bf16(ap1, bv1, ctx[j]);
    }
    __syncthreads();   // protect Kbuf/Vbuf before next TDM / staging pass
  }

  // ---- normalize + store [B,S,768] f32 -----------------------------------
  #pragma unroll
  for (int r = 0; r < 8; ++r) {
    const float inv = 1.0f / fmaxf(lstat[r], 1e-30f);
    const int s = q0 + r + half * 8;
    float* op = out + ((size_t)b * SQ + s) * HID + h * HD;
    #pragma unroll
    for (int j = 0; j < 4; ++j) op[j * 16 + ln] = ctx[j][r] * inv;
  }
}

// ---------------------------------------------------------------------------
extern "C" void kernel_launch(void* const* d_in, const int* in_sizes, int n_in,
                              void* d_out, int out_size, void* d_ws,
                              size_t ws_size, hipStream_t stream) {
  const float* hidden = (const float*)d_in[0];
  const float* amask  = (const float*)d_in[1];
  const float* rel    = (const float*)d_in[2];
  const float* rel2   = (const float*)d_in[3];
  const float* wq = (const float*)d_in[4]; const float* bq = (const float*)d_in[5];
  const float* wk = (const float*)d_in[6]; const float* bk = (const float*)d_in[7];
  const float* wv = (const float*)d_in[8]; const float* bv = (const float*)d_in[9];
  float* out = (float*)d_out;

  const size_t nHid = (size_t)NB * SQ * HID;        // 3,145,728
  const size_t nW   = (size_t)HID * HID;            // 589,824
  const size_t nHead = (size_t)NB * NH * SQ * HD;   // 3,145,728

  char* ws = (char*)d_ws;
  unsigned short* hidbf = (unsigned short*)ws;            ws += nHid * 2;
  unsigned short* wqT   = (unsigned short*)ws;            ws += nW * 2;
  unsigned short* wkT   = (unsigned short*)ws;            ws += nW * 2;
  unsigned short* wvT   = (unsigned short*)ws;            ws += nW * 2;
  unsigned short* qbf   = (unsigned short*)ws;            ws += nHead * 2;
  unsigned short* kbf   = (unsigned short*)ws;            ws += nHead * 2;
  unsigned short* vtb   = (unsigned short*)ws;            // ~28.7 MB total

  cvt_hidden<<<(int)((nHid + 255) / 256), 256, 0, stream>>>(hidden, hidbf,
                                                            (int)nHid);
  dim3 tg(HID / 32, HID / 32);
  cvt_w_tr<<<tg, 256, 0, stream>>>(wq, wqT);
  cvt_w_tr<<<tg, 256, 0, stream>>>(wk, wkT);
  cvt_w_tr<<<tg, 256, 0, stream>>>(wv, wvT);

  dim3 gq(NH, (NB * SQ) / (16 * 8), 3);   // (12, 32, 3)
  qkv_kernel<<<gq, 256, 0, stream>>>(hidbf, wqT, wkT, wvT, bq, bk, bv,
                                     qbf, kbf, vtb);

  dim3 ga(SQ / 64, NH, NB);               // (32, 12, 2)
  attn_kernel<<<ga, 128, 0, stream>>>(qbf, kbf, vtb, rel, rel2, amask, out);
}
